// GATv2_DX_DG_DR_77747497992605
// MI455X (gfx1250) — matrix operand order repaired and checked by reading, not compiled, not run on hardware
//
#include <hip/hip_runtime.h>
#include <math.h>

// ---------------------------------------------------------------------------
// GATv2 (2 layers) + attentional pooling, fused pipeline for gfx1250 (MI455X).
// All matmuls run through v_wmma_f32_16x16x32_f16 (wave32 WMMA, f32 accum).
// Weights / node features are pre-converted to f16 once, so the GEMM inner
// loop is pure b128 loads + WMMA (no per-fragment cvt chains).
// Segment softmax uses single-atomic float max (int trick) + f32 atomic adds
// that stay resident in the 192MB L2 (xl/xr/h are ~20MB each).
// ---------------------------------------------------------------------------

#define HID   256
#define ED    32
#define GDIM  256
#define RDIM  256
#define FDIM  (HID + GDIM + RDIM)   // 768
#define NEG_SLOPE 0.2f

typedef __attribute__((ext_vector_type(16))) _Float16 v16h;
typedef __attribute__((ext_vector_type(8)))  _Float16 v8h;
typedef __attribute__((ext_vector_type(8)))  float    v8f;

__device__ __forceinline__ void atomicMaxF(float* addr, float v) {
  // monotone int trick: works with -inf initialization, one atomic, no CAS loop
  if (v >= 0.0f) atomicMax((int*)addr, __float_as_int(v));
  else           atomicMin((unsigned int*)addr, (unsigned int)__float_as_int(v));
}

// ---------------------------------------------------------------------------
__global__ void fill_kernel(float* __restrict__ p, int n, float v) {
  int i = blockIdx.x * blockDim.x + threadIdx.x;
  if (i < n) p[i] = v;
}

__global__ void cvt_f16_kernel(const float* __restrict__ src,
                               _Float16* __restrict__ dst, int n) {
  int i = blockIdx.x * blockDim.x + threadIdx.x;
  if (i < n) dst[i] = (_Float16)src[i];
}

// ---- self-loop attr = mean of incoming edge attrs --------------------------
__global__ void loopattr_sum_kernel(const int* __restrict__ ei,
                                    const float* __restrict__ eattr,
                                    float* __restrict__ cnt,
                                    float* __restrict__ lsum, int E) {
  int t = blockIdx.x * blockDim.x + threadIdx.x;
  if (t >= E * ED) return;
  int e = t >> 5;              // ED == 32
  int k = t & 31;
  int d = ei[E + e];
  atomicAdd(&lsum[(size_t)d * ED + k], eattr[t]);
  if (k == 0) atomicAdd(&cnt[d], 1.0f);
}

__global__ void loopattr_div_kernel(float* __restrict__ lsum,
                                    const float* __restrict__ cnt, int N) {
  int t = blockIdx.x * blockDim.x + threadIdx.x;
  if (t >= N * ED) return;
  float c = cnt[t >> 5];
  if (c < 1.0f) c = 1.0f;
  lsum[t] *= (1.0f / c);
}

// ---- node transform: Y[N][256] = X[N][K] @ W[256][K]^T + bias --------------
__global__ __launch_bounds__(256) void gemm_nodes_kernel(
    const _Float16* __restrict__ X, const _Float16* __restrict__ W,
    const float* __restrict__ bias, float* __restrict__ Y, int N, int K) {
  const int lane   = threadIdx.x & 31;
  const int gw     = (blockIdx.x * blockDim.x + threadIdx.x) >> 5;
  const int rowTile = (gw >> 2) * 16;
  const int colBase = (gw & 3) * 64;
  if (rowTile >= N) return;
  const int colin = lane & 15;
  const int hlf   = lane >> 4;
  const int kA    = hlf * 16;    // A frag K pattern: 0-15 vs 16-31
  const int kB    = hlf * 16;    // B frag K pattern: 0-15 vs 16-31

  v8f acc[4] = {};
  for (int k0 = 0; k0 < K; k0 += 32) {
    const _Float16* pa = X + (size_t)(rowTile + colin) * K + k0 + kA;
    v8h a0 = *(const v8h*)pa;            // 16B aligned
    v8h a1 = *(const v8h*)(pa + 8);
    v16h a;
#pragma unroll
    for (int i = 0; i < 8; ++i) { a[i] = a0[i]; a[8 + i] = a1[i]; }
#pragma unroll
    for (int t = 0; t < 4; ++t) {
      const _Float16* pb = W + (size_t)(colBase + 16 * t + colin) * K + k0 + kB;
      v16h b = *(const v16h*)pb;         // 32B aligned, contiguous K span
      acc[t] = __builtin_amdgcn_wmma_f32_16x16x32_f16(
          false, a, false, b, (short)0, acc[t], false, false);
    }
  }
  const int rofs = hlf * 8;
#pragma unroll
  for (int t = 0; t < 4; ++t) {
    int col = colBase + 16 * t + colin;
    float bv = bias[col];
#pragma unroll
    for (int j = 0; j < 8; ++j)
      Y[(size_t)(rowTile + rofs + j) * HID + col] = acc[t][j] + bv;
  }
}

// ---- pass 1: edge logits + per-dst max -------------------------------------
__global__ __launch_bounds__(256) void edge_logit_kernel(
    const int* __restrict__ ei, const float* __restrict__ eattr,
    const float* __restrict__ lattr, const float* __restrict__ We,
    const float* __restrict__ att, const float* __restrict__ xl,
    const float* __restrict__ xr, float* __restrict__ logit,
    float* __restrict__ mmax, int E, int N, int numTiles) {
  __shared__ _Float16 sWe[HID * ED];          // [col][k], 16 KB
  for (int i = threadIdx.x; i < HID * ED; i += blockDim.x)
    sWe[i] = (_Float16)We[i];
  __syncthreads();

  const int lane  = threadIdx.x & 31;
  const int wid   = threadIdx.x >> 5;
  const int wpb   = blockDim.x >> 5;
  const int colin = lane & 15;
  const int hlf   = lane >> 4;
  const int rofs  = hlf * 8;
  const int kA    = hlf * 16;
  const int kB    = hlf * 16;

  for (int tile = blockIdx.x * wpb + wid; tile < numTiles;
       tile += gridDim.x * wpb) {
    int item = tile * 16 + colin;
    int s, d;
    const float* pa;
    if (item < E) {
      s = ei[item]; d = ei[E + item];
      pa = eattr + (size_t)item * ED;
    } else {
      s = item - E; d = s;
      pa = lattr + (size_t)(item - E) * ED;
    }
    v16h a;
#pragma unroll
    for (int i = 0; i < 8; ++i) a[i]     = (_Float16)pa[kA + i];
#pragma unroll
    for (int i = 0; i < 8; ++i) a[8 + i] = (_Float16)pa[kA + 8 + i];

    int srow[8], drow[8];
#pragma unroll
    for (int j = 0; j < 8; ++j) {
      srow[j] = __shfl(s, rofs + j, 32);
      drow[j] = __shfl(d, rofs + j, 32);
    }

    float lacc[8];
#pragma unroll
    for (int j = 0; j < 8; ++j) lacc[j] = 0.0f;

#pragma unroll 4
    for (int t = 0; t < 16; ++t) {          // 16 column tiles -> HID=256
      int col = t * 16 + colin;
      v16h b = *(const v16h*)(&sWe[col * ED + kB]);   // 32B-aligned LDS read
      v8f c = {};
      c = __builtin_amdgcn_wmma_f32_16x16x32_f16(
          false, a, false, b, (short)0, c, false, false);
      float av = att[col];
#pragma unroll
      for (int j = 0; j < 8; ++j) {
        float v = c[j] + xl[(size_t)srow[j] * HID + col]
                       + xr[(size_t)drow[j] * HID + col];
        v = v > 0.0f ? v : NEG_SLOPE * v;
        lacc[j] += v * av;
      }
    }
    float myv = 0.0f;
#pragma unroll
    for (int j = 0; j < 8; ++j) {
      float v = lacc[j];
      v += __shfl_xor(v, 1, 32);
      v += __shfl_xor(v, 2, 32);
      v += __shfl_xor(v, 4, 32);
      v += __shfl_xor(v, 8, 32);
      if (colin == j) myv = v;
    }
    if (colin < 8) {
      int r = rofs + colin;
      int item_r = tile * 16 + r;
      int dr = __shfl(d, r, 32);
      logit[item_r] = myv;
      atomicMaxF(&mmax[dr], myv);
    }
  }
}

// ---- pass 2: exp + per-dst denom -------------------------------------------
__global__ void softmax_ex_kernel(const int* __restrict__ ei,
                                  const float* __restrict__ logit,
                                  const float* __restrict__ mmax,
                                  float* __restrict__ expv,
                                  float* __restrict__ denom, int E, int T) {
  int i = blockIdx.x * blockDim.x + threadIdx.x;
  if (i >= T) return;
  int d = (i < E) ? ei[E + i] : (i - E);
  float ex = __expf(logit[i] - mmax[d]);
  expv[i] = ex;
  atomicAdd(&denom[d], ex);
}

// ---- pass 3: out[d] += alpha * xl[s] ---------------------------------------
__global__ void aggregate_kernel(const int* __restrict__ ei,
                                 const float* __restrict__ expv,
                                 const float* __restrict__ denom,
                                 const float* __restrict__ xl,
                                 float* __restrict__ acc, int E, int T) {
  int gw   = (blockIdx.x * blockDim.x + threadIdx.x) >> 5;
  int lane = threadIdx.x & 31;
  if (gw >= T) return;
  int s, d;
  if (gw < E) { s = ei[gw]; d = ei[E + gw]; } else { s = gw - E; d = s; }
  float alpha = expv[gw] / denom[d];
  const float* xs = xl + (size_t)s * HID;
  float* po = acc + (size_t)d * HID;
#pragma unroll
  for (int c = 0; c < HID; c += 32)
    atomicAdd(&po[c + lane], alpha * xs[c + lane]);
}

// ---- bias + ELU (in place), optional f16 mirror for the next GEMM ----------
__global__ void bias_elu_kernel(float* __restrict__ h,
                                const float* __restrict__ bias,
                                _Float16* __restrict__ h16, int total) {
  int i = blockIdx.x * blockDim.x + threadIdx.x;
  if (i >= total) return;
  float v = h[i] + bias[i & (HID - 1)];
  v = v > 0.0f ? v : (__expf(v) - 1.0f);
  h[i] = v;
  if (h16) h16[i] = (_Float16)v;
}

// ---- pooling gate ----------------------------------------------------------
__global__ void gate_kernel(const float* __restrict__ h,
                            const float* __restrict__ Wg,
                            const float* __restrict__ bg,
                            const int* __restrict__ batch,
                            float* __restrict__ gate,
                            float* __restrict__ gmax, int N) {
  int gw   = (blockIdx.x * blockDim.x + threadIdx.x) >> 5;
  int lane = threadIdx.x & 31;
  if (gw >= N) return;
  const float* hn = h + (size_t)gw * HID;
  float acc = 0.0f;
#pragma unroll
  for (int c = 0; c < HID; c += 32) acc += hn[c + lane] * Wg[c + lane];
  acc += __shfl_xor(acc, 16, 32);
  acc += __shfl_xor(acc, 8, 32);
  acc += __shfl_xor(acc, 4, 32);
  acc += __shfl_xor(acc, 2, 32);
  acc += __shfl_xor(acc, 1, 32);
  if (lane == 0) {
    float v = acc + bg[0];
    gate[gw] = v;
    atomicMaxF(&gmax[batch[gw]], v);
  }
}

__global__ void gate_ex_kernel(const float* __restrict__ gate,
                               const float* __restrict__ gmax,
                               const int* __restrict__ batch,
                               float* __restrict__ gex,
                               float* __restrict__ gden, int N) {
  int i = blockIdx.x * blockDim.x + threadIdx.x;
  if (i >= N) return;
  float ex = __expf(gate[i] - gmax[batch[i]]);
  gex[i] = ex;
  atomicAdd(&gden[batch[i]], ex);
}

__global__ void pool_kernel(const float* __restrict__ h,
                            const float* __restrict__ gex,
                            const float* __restrict__ gden,
                            const int* __restrict__ batch,
                            float* __restrict__ pooled, int N) {
  int gw   = (blockIdx.x * blockDim.x + threadIdx.x) >> 5;
  int lane = threadIdx.x & 31;
  if (gw >= N) return;
  int b = batch[gw];
  float a = gex[gw] / gden[b];
  const float* hn = h + (size_t)gw * HID;
  float* po = pooled + (size_t)b * HID;
#pragma unroll
  for (int c = 0; c < HID; c += 32)
    atomicAdd(&po[c + lane], a * hn[c + lane]);
}

// ---- final: out[b] = [pooled|g|r] . Wf + bf --------------------------------
__global__ void final_kernel(const float* __restrict__ pooled,
                             const float* __restrict__ g,
                             const float* __restrict__ r,
                             const float* __restrict__ Wf,
                             const float* __restrict__ bf,
                             float* __restrict__ out, int B) {
  int gw   = (blockIdx.x * blockDim.x + threadIdx.x) >> 5;
  int lane = threadIdx.x & 31;
  if (gw >= B) return;
  float acc = 0.0f;
#pragma unroll
  for (int i0 = 0; i0 < FDIM; i0 += 32) {
    int i = i0 + lane;
    float v;
    if (i < HID)              v = pooled[(size_t)gw * HID + i];
    else if (i < HID + GDIM)  v = g[(size_t)gw * GDIM + (i - HID)];
    else                      v = r[(size_t)gw * RDIM + (i - HID - GDIM)];
    acc += v * Wf[i];
  }
  acc += __shfl_xor(acc, 16, 32);
  acc += __shfl_xor(acc, 8, 32);
  acc += __shfl_xor(acc, 4, 32);
  acc += __shfl_xor(acc, 2, 32);
  acc += __shfl_xor(acc, 1, 32);
  if (lane == 0) out[gw] = acc + bf[0];
}

// ===========================================================================
extern "C" void kernel_launch(void* const* d_in, const int* in_sizes, int n_in,
                              void* d_out, int out_size, void* d_ws,
                              size_t ws_size, hipStream_t stream) {
  const float* x     = (const float*)d_in[0];
  const int*   ei    = (const int*)d_in[1];
  const float* eattr = (const float*)d_in[2];
  const int*   batch = (const int*)d_in[3];
  const float* g     = (const float*)d_in[4];
  const float* r     = (const float*)d_in[5];
  const float* W1l = (const float*)d_in[6];  const float* b1l = (const float*)d_in[7];
  const float* W1r = (const float*)d_in[8];  const float* b1r = (const float*)d_in[9];
  const float* W1e = (const float*)d_in[10]; const float* att1 = (const float*)d_in[11];
  const float* bias1 = (const float*)d_in[12];
  const float* W2l = (const float*)d_in[13]; const float* b2l = (const float*)d_in[14];
  const float* W2r = (const float*)d_in[15]; const float* b2r = (const float*)d_in[16];
  const float* W2e = (const float*)d_in[17]; const float* att2 = (const float*)d_in[18];
  const float* bias2 = (const float*)d_in[19];
  const float* Wg = (const float*)d_in[20]; const float* bg = (const float*)d_in[21];
  const float* Wf = (const float*)d_in[22]; const float* bf = (const float*)d_in[23];

  const int XD = 128;
  const int N = in_sizes[0] / XD;     // 20000 (multiple of 16)
  const int E = in_sizes[2] / ED;     // 320000
  const int B = in_sizes[4] / GDIM;   // 512
  const int T = E + N;                // 340000 (multiple of 16)

  // ---- workspace layout (all regions stay 32B aligned) ----
  float* w = (float*)d_ws;
  float* CNT   = w; w += N;
  float* LATTR = w; w += (size_t)N * ED;
  float* XL    = w; w += (size_t)N * HID;
  float* XR    = w; w += (size_t)N * HID;
  float* ACC   = w; w += (size_t)N * HID;   // h1, then re-zeroed for h2
  float* LOGIT = w; w += T;
  float* EXPV  = w; w += T;
  float* MMAX  = w; w += N;
  float* DEN   = w; w += N;
  float* GATE  = w; w += N;
  float* GEX   = w; w += N;
  float* GMAX  = w; w += B;
  float* GDEN  = w; w += B;
  float* POOL  = w; w += (size_t)B * HID;
  _Float16* hw = (_Float16*)w;
  _Float16* XH16  = hw; hw += (size_t)N * HID;    // x (N*128) then h1 (N*256)
  _Float16* W1L16 = hw; hw += (size_t)HID * XD;
  _Float16* W1R16 = hw; hw += (size_t)HID * XD;
  _Float16* W2L16 = hw; hw += (size_t)HID * HID;
  _Float16* W2R16 = hw; hw += (size_t)HID * HID;

  auto fill = [&](float* p, int n, float v) {
    fill_kernel<<<(n + 255) / 256, 256, 0, stream>>>(p, n, v);
  };
  auto cvt = [&](const float* src, _Float16* dst, int n) {
    cvt_f16_kernel<<<(n + 255) / 256, 256, 0, stream>>>(src, dst, n);
  };

  const int gemmBlocks = ((N / 16) * 4 + 7) / 8;
  const int numTiles   = T / 16;
  const int edgeBlocks = (numTiles + 7) / 8;
  const int waveBlocksT = (T * 32 + 255) / 256;
  const int waveBlocksN = (N * 32 + 255) / 256;
  const float NEG_INF = -INFINITY;

  // pre-convert weights + input features to f16 (hoists cvt out of GEMMs)
  cvt(x,   XH16,  N * XD);
  cvt(W1l, W1L16, HID * XD);
  cvt(W1r, W1R16, HID * XD);
  cvt(W2l, W2L16, HID * HID);
  cvt(W2r, W2R16, HID * HID);

  // self-loop edge attr (same for both layers)
  fill(CNT, N, 0.0f);
  fill(LATTR, N * ED, 0.0f);
  loopattr_sum_kernel<<<(E * ED + 255) / 256, 256, 0, stream>>>(ei, eattr, CNT, LATTR, E);
  loopattr_div_kernel<<<(N * ED + 255) / 256, 256, 0, stream>>>(LATTR, CNT, N);

  // ---------------- layer 1 ----------------
  gemm_nodes_kernel<<<gemmBlocks, 256, 0, stream>>>(XH16, W1L16, b1l, XL, N, XD);
  gemm_nodes_kernel<<<gemmBlocks, 256, 0, stream>>>(XH16, W1R16, b1r, XR, N, XD);
  fill(MMAX, N, NEG_INF);
  fill(DEN, N, 0.0f);
  fill(ACC, N * HID, 0.0f);
  edge_logit_kernel<<<edgeBlocks, 256, 0, stream>>>(ei, eattr, LATTR, W1e, att1,
                                                    XL, XR, LOGIT, MMAX, E, N, numTiles);
  softmax_ex_kernel<<<(T + 255) / 256, 256, 0, stream>>>(ei, LOGIT, MMAX, EXPV, DEN, E, T);
  aggregate_kernel<<<waveBlocksT, 256, 0, stream>>>(ei, EXPV, DEN, XL, ACC, E, T);
  bias_elu_kernel<<<(N * HID + 255) / 256, 256, 0, stream>>>(ACC, bias1, XH16, N * HID);

  // ---------------- layer 2 ----------------
  gemm_nodes_kernel<<<gemmBlocks, 256, 0, stream>>>(XH16, W2L16, b2l, XL, N, HID);
  gemm_nodes_kernel<<<gemmBlocks, 256, 0, stream>>>(XH16, W2R16, b2r, XR, N, HID);
  fill(MMAX, N, NEG_INF);
  fill(DEN, N, 0.0f);
  fill(ACC, N * HID, 0.0f);
  edge_logit_kernel<<<edgeBlocks, 256, 0, stream>>>(ei, eattr, LATTR, W2e, att2,
                                                    XL, XR, LOGIT, MMAX, E, N, numTiles);
  softmax_ex_kernel<<<(T + 255) / 256, 256, 0, stream>>>(ei, LOGIT, MMAX, EXPV, DEN, E, T);
  aggregate_kernel<<<waveBlocksT, 256, 0, stream>>>(ei, EXPV, DEN, XL, ACC, E, T);
  bias_elu_kernel<<<(N * HID + 255) / 256, 256, 0, stream>>>(ACC, bias2, (_Float16*)nullptr, N * HID);

  // ---------------- attentional pooling + final ----------------
  fill(GMAX, B, NEG_INF);
  fill(GDEN, B, 0.0f);
  fill(POOL, B * HID, 0.0f);
  gate_kernel<<<waveBlocksN, 256, 0, stream>>>(ACC, Wg, bg, batch, GATE, GMAX, N);
  gate_ex_kernel<<<(N + 255) / 256, 256, 0, stream>>>(GATE, GMAX, batch, GEX, GDEN, N);
  pool_kernel<<<waveBlocksN, 256, 0, stream>>>(ACC, GEX, GDEN, batch, POOL, N);
  final_kernel<<<(B * 32 + 255) / 256, 256, 0, stream>>>(POOL, g, r, Wf, bf,
                                                         (float*)d_out, B);
}
